// ConnectionV2_85255100825969
// MI455X (gfx1250) — 16-bit and 8-bit matrix operand order repaired and checked by reading, compiled, not run on hardware
//
#include <hip/hip_runtime.h>
#include <hip/hip_bf16.h>
#include <math.h>

#define BQ    4096
#define DQ    32
#define HID   256
#define NSTEP 5
#define NCOL  1024
#define KTOT  8192
#define NT_W3 64
#define KT64  128
#define MROWS 32

#define SCALE_A 0x1p27f
#define SCALE_B 0x1p13f
#define DESCALE 0x1p-40f

typedef __bf16 bf16_t;
typedef bf16_t v16bf __attribute__((ext_vector_type(16)));
typedef bf16_t v8bf  __attribute__((ext_vector_type(8)));
typedef float  v8f   __attribute__((ext_vector_type(8)));
typedef int    v8i   __attribute__((ext_vector_type(8)));

static __device__ __forceinline__ v8f wmma_bf16(v16bf a, v16bf b, v8f c) {
  return __builtin_amdgcn_wmma_f32_16x16x32_bf16(false, a, false, b, (short)0, c,
                                                 false, false);
}
static __device__ __forceinline__ v8f wmma_fp8(v8i a, v8i b, v8f c) {
  return __builtin_amdgcn_wmma_f32_16x16x64_fp8_fp8(a, b, (short)0, c, false, false);
}

static __device__ __forceinline__ unsigned f32_to_e4m3(float x) {
  union { float f; unsigned u; } c; c.f = x;
  unsigned s = (c.u >> 24) & 0x80u;
  float ax = fabsf(x);
  if (ax < 0.0009765625f) return s;
  if (ax >= 448.0f) return s | 0x7Eu;
  if (ax < 0.015625f) {
    int q = (int)(ax * 512.0f + 0.5f);
    return s | (unsigned)(q > 7 ? 8 : q);
  }
  unsigned u = (c.u & 0x7FFFFFFFu) + 0x00080000u;
  int exp8 = (int)(u >> 23) - 127;
  unsigned m3 = (u >> 20) & 7u;
  unsigned bits = (unsigned)((exp8 + 7) << 3) | m3;
  if (bits >= 0x7Fu) bits = 0x7Eu;
  return s | bits;
}

__global__ void pack_w1_kernel(const float* __restrict__ W1, bf16_t* __restrict__ out) {
  int o = blockIdx.x * 256 + threadIdx.x;
  int e = o & 15, L = (o >> 4) & 31, nt = o >> 9;
  int K = e + ((L & 16) ? 16 : 0);
  int n = nt * 16 + (L & 15);
  out[o] = (bf16_t)W1[K * HID + n];
}

__global__ void pack_w2_kernel(const float* __restrict__ W2, bf16_t* __restrict__ out) {
  int o = blockIdx.x * 256 + threadIdx.x;
  int e = o & 15, L = (o >> 4) & 31;
  int nt = (o >> 9) & 15, kt = o >> 13;
  int K = kt * 32 + e + ((L & 16) ? 16 : 0);
  int n = nt * 16 + (L & 15);
  out[o] = (bf16_t)W2[K * HID + n];
}

__global__ void pack_w3_fp8_kernel(const float* __restrict__ W3,
                                   unsigned char* __restrict__ out) {
  int o = blockIdx.x * 256 + threadIdx.x;
  int e    = o & 31;
  int L    = (o >> 5) & 31;
  int nt   = (o >> 10) & 63;
  int kt64 = o >> 16;
  int K = kt64 * 64 + 32 * (e >> 4) + ((L & 16) ? 16 : 0) + (e & 15);
  int n = nt * 16 + (L & 15);
  int k = K >> 5, m = K & 31;
  int i = n >> 5, j = n & 31;
  float a = W3[(size_t)k * 32768 + m * 1024 + i * 32 + j];
  float b = W3[(size_t)k * 32768 + m * 1024 + j * 32 + i];
  out[o] = (unsigned char)f32_to_e4m3(0.5f * (a - b) * SCALE_B);
}

__global__ void db3_kernel(const float* __restrict__ q_from, const float* __restrict__ q_to,
                           const float* __restrict__ b3, bf16_t* __restrict__ db3) {
  int o = blockIdx.x * 256 + threadIdx.x;
  int n = o & 1023, b = o >> 10;
  int i = n >> 5, j = n & 31;
  float s = 0.f;
  for (int m = 0; m < 32; ++m) {
    float dqm = (q_to[b * 32 + m] - q_from[b * 32 + m]) * (1.0f / NSTEP);
    s += dqm * 0.5f * (b3[m * 1024 + i * 32 + j] - b3[m * 1024 + j * 32 + i]);
  }
  db3[o] = (bf16_t)s;
}

__global__ __launch_bounds__(256) void pack_p_kernel(
    const bf16_t* __restrict__ h_ws,
    const float* __restrict__ q_from, const float* __restrict__ q_to,
    unsigned* __restrict__ Pf) {
  __shared__ float dqlds[16 * 32];
  const int tid = threadIdx.x;
  const int st  = blockIdx.x >> 7;
  for (int idx = tid; idx < 512; idx += 256) {
    int g = (st * 16) * 32 + idx;
    dqlds[idx] = (q_to[g] - q_from[g]) * (1.0f / NSTEP);
  }
  __syncthreads();
  int o = blockIdx.x * 256 + tid;
  int d    = o & 7;
  int L    = (o >> 3) & 31;
  int kt64 = (o >> 8) & 127;
  int rowL = L & 15;
  int hi8  = (L & 16) ? 8 : 0;
  int row_g = st * 16 + rowL;
  unsigned dw = 0;
  #pragma unroll
  for (int b4 = 0; b4 < 4; ++b4) {
    int e = d * 4 + b4;
    int K = kt64 * 64 + 32 * (e >> 4) + (e & 15) + 2 * hi8;
    int k_h = K >> 5, m = K & 31;
    float hv = (float)h_ws[(size_t)row_g * 256 + k_h];
    float val = hv * dqlds[rowL * 32 + m] * SCALE_A;
    dw |= f32_to_e4m3(val) << (8 * b4);
  }
  Pf[o] = dw;
}

__global__ __launch_bounds__(256) void mlp_kernel(
    const float* __restrict__ q_from, const float* __restrict__ q_to, int step,
    const bf16_t* __restrict__ W1f, const float* __restrict__ b1,
    const bf16_t* __restrict__ W2f, const float* __restrict__ b2,
    bf16_t* __restrict__ h_out)
{
  __shared__ __align__(16) bf16_t qlds[16 * 32];
  __shared__ __align__(16) bf16_t h1[16 * 256];

  const int tid  = threadIdx.x;
  const int lane = tid & 31;
  const int w    = tid >> 5;
  const int bBase = blockIdx.x * 16;

  for (int idx = tid; idx < 16 * 32; idx += 256) {
    float q0 = q_from[bBase * DQ + idx];
    float dq = (q_to[bBase * DQ + idx] - q0) * (1.0f / NSTEP);
    qlds[idx] = (bf16_t)(q0 + (float)step * dq);
  }
  __syncthreads();

  const int row    = lane & 15;
  const int hiHalf = lane >> 4;
  const int abase  = hiHalf * 16;

  v8bf qlo = *(const v8bf*)&qlds[row * 32 + abase];
  v8bf qhi = *(const v8bf*)&qlds[row * 32 + abase + 8];
  v16bf aq = __builtin_shufflevector(qlo, qhi, 0,1,2,3,4,5,6,7,8,9,10,11,12,13,14,15);

  #pragma unroll
  for (int t = 0; t < 2; ++t) {
    int nt = w * 2 + t;
    v16bf bf = *(const v16bf*)&W1f[(nt * 32 + lane) * 16];
    v8f acc = {0.f,0.f,0.f,0.f,0.f,0.f,0.f,0.f};
    acc = wmma_bf16(aq, bf, acc);
    int n = nt * 16 + (lane & 15);
    float bias = b1[n];
    #pragma unroll
    for (int r8 = 0; r8 < 8; ++r8) {
      int M = r8 + hiHalf * 8;
      h1[M * 256 + n] = (bf16_t)tanhf(acc[r8] + bias);
    }
  }
  __syncthreads();

  v8f acc2[2];
  #pragma unroll
  for (int t = 0; t < 2; ++t) acc2[t] = (v8f){0.f,0.f,0.f,0.f,0.f,0.f,0.f,0.f};
  for (int kt = 0; kt < 8; ++kt) {
    int kb = kt * 32 + abase;
    v8bf lo = *(const v8bf*)&h1[row * 256 + kb];
    v8bf hi = *(const v8bf*)&h1[row * 256 + kb + 8];
    v16bf a = __builtin_shufflevector(lo, hi, 0,1,2,3,4,5,6,7,8,9,10,11,12,13,14,15);
    #pragma unroll
    for (int t = 0; t < 2; ++t) {
      int nt = w * 2 + t;
      v16bf bf = *(const v16bf*)&W2f[((kt * 16 + nt) * 32 + lane) * 16];
      acc2[t] = wmma_bf16(a, bf, acc2[t]);
    }
  }
  #pragma unroll
  for (int t = 0; t < 2; ++t) {
    int n = (w * 2 + t) * 16 + (lane & 15);
    float bias = b2[n];
    #pragma unroll
    for (int r8 = 0; r8 < 8; ++r8) {
      int M = r8 + hiHalf * 8;
      h_out[(size_t)(bBase + M) * 256 + n] = (bf16_t)tanhf(acc2[t][r8] + bias);
    }
  }
}

__global__ __launch_bounds__(256) void transport_kernel(
    const unsigned char* __restrict__ Pf,
    const unsigned char* __restrict__ W3f8,
    const bf16_t* __restrict__ db3,
    const float* __restrict__ vin, float* __restrict__ vout)
{
  __shared__ float vold[MROWS * 32];
  __shared__ float Ac[16 * 1024];

  const int tid  = threadIdx.x;
  const int lane = tid & 31;
  const int w    = tid >> 5;
  const int bBase = blockIdx.x * MROWS;

  for (int idx = tid; idx < MROWS * 32; idx += 256)
    vold[idx] = vin[bBase * DQ + idx];
  __syncthreads();

  const int ntBase = w * 8;
  const unsigned char* pa0 = Pf + ((size_t)(blockIdx.x * 2) * KT64 * 32 + lane) * 32;
  const unsigned char* pa1 = pa0 + (size_t)KT64 * 32 * 32;
  const unsigned char* pb  = W3f8 + ((size_t)ntBase * 32 + lane) * 32;

  v8f acc0[8], acc1[8];
  #pragma unroll
  for (int t = 0; t < 8; ++t) {
    acc0[t] = (v8f){0.f,0.f,0.f,0.f,0.f,0.f,0.f,0.f};
    acc1[t] = (v8f){0.f,0.f,0.f,0.f,0.f,0.f,0.f,0.f};
  }

  for (int kt = 0; kt < KT64; ++kt) {
    v8i a0 = *(const v8i*)(pa0 + (size_t)kt * 1024);
    v8i a1 = *(const v8i*)(pa1 + (size_t)kt * 1024);
    const unsigned char* bpt = pb + (size_t)kt * 65536;
    #pragma unroll
    for (int t = 0; t < 8; ++t) {
      v8i bfrag = *(const v8i*)(bpt + (size_t)t * 1024);
      acc0[t] = wmma_fp8(a0, bfrag, acc0[t]);
      acc1[t] = wmma_fp8(a1, bfrag, acc1[t]);
    }
  }

  const int mb = (lane >> 4) * 8;
  #pragma unroll
  for (int phase = 0; phase < 2; ++phase) {
    const v8f* acc = (phase == 0) ? acc0 : acc1;
    #pragma unroll
    for (int t = 0; t < 8; ++t) {
      int col = (ntBase + t) * 16 + (lane & 15);
      #pragma unroll
      for (int r8 = 0; r8 < 8; ++r8)
        Ac[(r8 + mb) * 1024 + col] = acc[t][r8] * DESCALE;
    }
    __syncthreads();
    #pragma unroll
    for (int pp = 0; pp < 2; ++pp) {
      int p = tid + pp * 256;
      int b = p >> 5, i = p & 31;
      int brow = phase * 16 + b;
      const float*  arow = &Ac[b * 1024 + i * 32];
      const bf16_t* drow = &db3[(size_t)(bBase + brow) * 1024 + i * 32];
      const float*  vrow = &vold[brow * 32];
      float s = 0.f;
      #pragma unroll
      for (int j = 0; j < 32; ++j) s += (arow[j] + (float)drow[j]) * vrow[j];
      vout[(size_t)(bBase + brow) * DQ + i] = vrow[i] - s;
    }
    __syncthreads();
  }
}

extern "C" void kernel_launch(void* const* d_in, const int* in_sizes, int n_in,
                              void* d_out, int out_size, void* d_ws, size_t ws_size,
                              hipStream_t stream)
{
  (void)in_sizes; (void)n_in; (void)out_size; (void)ws_size;
  const float* q_from = (const float*)d_in[0];
  const float* q_to   = (const float*)d_in[1];
  const float* v      = (const float*)d_in[2];
  const float* W1     = (const float*)d_in[3];
  const float* b1     = (const float*)d_in[4];
  const float* W2     = (const float*)d_in[5];
  const float* b2     = (const float*)d_in[6];
  const float* W3     = (const float*)d_in[7];
  const float* b3     = (const float*)d_in[8];
  float* v_final      = (float*)d_out;

  char* ws = (char*)d_ws;
  unsigned char* W3f8 = (unsigned char*)ws; ws += (size_t)KTOT * NCOL;
  unsigned char* Pf_b = (unsigned char*)ws;
  unsigned*      Pf   = (unsigned*)ws;      ws += (size_t)BQ * KTOT;
  bf16_t*        db3  = (bf16_t*)ws;        ws += (size_t)BQ * NCOL * 2;
  bf16_t*        W1f  = (bf16_t*)ws;        ws += (size_t)DQ * HID * 2;
  bf16_t*        W2f  = (bf16_t*)ws;        ws += (size_t)HID * HID * 2;
  bf16_t*        h_ws = (bf16_t*)ws;        ws += (size_t)BQ * HID * 2;
  float*         v_ws = (float*)ws;         ws += (size_t)BQ * DQ * 4;

  pack_w3_fp8_kernel<<<(KTOT * NCOL) / 256, 256, 0, stream>>>(W3, W3f8);
  pack_w1_kernel<<<(DQ * HID) / 256, 256, 0, stream>>>(W1, W1f);
  pack_w2_kernel<<<(HID * HID) / 256, 256, 0, stream>>>(W2, W2f);
  db3_kernel<<<(BQ * NCOL) / 256, 256, 0, stream>>>(q_from, q_to, b3, db3);

  for (int s = 0; s < NSTEP; ++s) {
    mlp_kernel<<<BQ / 16, 256, 0, stream>>>(q_from, q_to, s, W1f, b1, W2f, b2, h_ws);
    pack_p_kernel<<<(BQ * KTOT / 4) / 256, 256, 0, stream>>>(h_ws, q_from, q_to, Pf);
    const float* vi = (s == 0) ? v : v_ws;
    float* vo = (s == NSTEP - 1) ? v_final : v_ws;
    transport_kernel<<<BQ / MROWS, 256, 0, stream>>>(Pf_b, W3f8, db3, vi, vo);
  }
}
